// PoissonPinn_76647986364581
// MI455X (gfx1250) — compile-verified, matrix operand order checked by reading, not run on hardware
//
#include <hip/hip_runtime.h>

typedef _Float16 half_t;
typedef __attribute__((ext_vector_type(16))) _Float16 v16h;
typedef __attribute__((ext_vector_type(8)))  _Float16 v8h;
typedef __attribute__((ext_vector_type(8)))  float    v8f;

union AFrag { v16h v; v8h h[2]; };

__device__ __forceinline__ float fast_tanh(float z) {
    // tanh(z) = 1 - 2/(exp(2z)+1); robust at +/-inf
    float e = __expf(2.0f * z);
    return 1.0f - 2.0f * __builtin_amdgcn_rcpf(e + 1.0f);
}

constexpr int HID     = 256;     // hidden width
constexpr int PTS     = 64;      // points per block (4 waves x 16)
constexpr int THREADS = 128;     // 4 wave32
constexpr int ROWP    = HID + 8; // LDS row pitch in halfs (+16B to break bank conflicts)
constexpr int NBT     = 4;       // N-tiles register-blocked per pass

// ---------------------------------------------------------------------------
// Pre-pass: W1..W3 (256x256 f32, [K][N]) -> WT f16 [layer][N][K] in workspace
// ---------------------------------------------------------------------------
__global__ void wt_transpose_kernel(const float* __restrict__ W1,
                                    const float* __restrict__ W2,
                                    const float* __restrict__ W3,
                                    half_t* __restrict__ WT) {
    int l = blockIdx.y;
    const float* W = (l == 0) ? W1 : ((l == 1) ? W2 : W3);
    int idx = blockIdx.x * blockDim.x + threadIdx.x;   // 0..65535
    int k = idx >> 8, n = idx & 255;
    WT[l * HID * HID + n * HID + k] = (half_t)W[k * HID + n];
}

// ---------------------------------------------------------------------------
// Equation kernel: forward-mode propagation of (h, h', h'') -> d2u/dx2.
// Each wave owns 16 points; activations live in LDS ping-pong (wave-private
// rows => no inter-layer barriers). Weights stream from L2 as f16 W^T rows.
// 4 N-tiles register-blocked: 12 independent WMMAs per K-step, A-frags
// loaded once per K-step and reused from registers across the 4 tiles.
// ---------------------------------------------------------------------------
__global__ __launch_bounds__(THREADS) void pinn_eq_kernel(
    const float* __restrict__ x,
    const half_t* __restrict__ WT,
    const float* __restrict__ W0,
    const float* __restrict__ b0,
    const float* __restrict__ b1,
    const float* __restrict__ b2,
    const float* __restrict__ b3,
    const float* __restrict__ W4,
    float* __restrict__ out) {
    extern __shared__ char smem_raw[];
    half_t* act = (half_t*)smem_raw;           // [2 buf][3 mat][PTS][ROWP]
    const int MATS = 3 * PTS * ROWP;           // one buffer's span

    const int tid   = threadIdx.x;
    const int pbase = blockIdx.x * PTS;

    // ---- layer 0 (scalar input, analytic seeds) -> buffer 0 ----
    for (int e = tid; e < PTS * HID; e += THREADS) {
        int p = e >> 8, c = e & 255;
        float xv = x[pbase + p];
        float w  = W0[c];
        float z  = xv * w + b0[c];
        float t  = fast_tanh(z);
        float s  = 1.0f - t * t;
        act[(0 * PTS + p) * ROWP + c] = (half_t)t;                          // h
        act[(1 * PTS + p) * ROWP + c] = (half_t)(s * w);                    // h'
        act[(2 * PTS + p) * ROWP + c] = (half_t)(-2.0f * t * s * w * w);    // h''
    }
    __syncthreads();

    const int wid   = tid >> 5;
    const int lane  = tid & 31;
    const int idx16 = lane & 15;    // A: row-in-tile; B/C/D: column-in-tile
    const int hi    = lane >> 4;    // upper/lower K-half selector
    const int p0    = wid * 16;     // this wave's point-row base

    #pragma unroll
    for (int l = 0; l < 3; ++l) {
        half_t* rbuf = act + (l & 1) * MATS;
        half_t* wbuf = act + ((l & 1) ^ 1) * MATS;
        const half_t* wt = WT + l * HID * HID;
        const float*  bl = (l == 0) ? b1 : ((l == 1) ? b2 : b3);

        for (int ng = 0; ng < 16 / NBT; ++ng) {
            // Warm the WGP cache for the next N-group's W^T rows.
            if (ng < 16 / NBT - 1) {
                #pragma unroll
                for (int t = 0; t < NBT; ++t) {
                    const int ncol = ((ng + 1) * NBT + t) * 16 + idx16;
                    __builtin_prefetch(wt + ncol * HID + hi * 8, 0, 3);
                }
            }

            v8f az[NBT], ad[NBT], a2v[NBT];
            #pragma unroll
            for (int t = 0; t < NBT; ++t) {
                az[t]  = (v8f){0,0,0,0,0,0,0,0};
                ad[t]  = (v8f){0,0,0,0,0,0,0,0};
                a2v[t] = (v8f){0,0,0,0,0,0,0,0};
            }

            for (int kb = 0; kb < 8; ++kb) {
                const int ko = kb * 32 + hi * 8;   // K offset for this lane half
                AFrag a0, a1, a2;
                const half_t* ar = rbuf + (p0 + idx16) * ROWP + ko;
                a0.h[0] = *(const v8h*)(ar);
                a0.h[1] = *(const v8h*)(ar + 16);
                a1.h[0] = *(const v8h*)(ar + PTS * ROWP);
                a1.h[1] = *(const v8h*)(ar + PTS * ROWP + 16);
                a2.h[0] = *(const v8h*)(ar + 2 * PTS * ROWP);
                a2.h[1] = *(const v8h*)(ar + 2 * PTS * ROWP + 16);
                #pragma unroll
                for (int t = 0; t < NBT; ++t) {
                    const int col = (ng * NBT + t) * 16 + idx16;
                    AFrag bf;
                    const half_t* br = wt + col * HID + ko;  // W^T row: contiguous K
                    bf.h[0] = *(const v8h*)(br);
                    bf.h[1] = *(const v8h*)(br + 16);
                    az[t]  = __builtin_amdgcn_wmma_f32_16x16x32_f16(false, a0.v, false, bf.v, (short)0, az[t],  false, false);
                    ad[t]  = __builtin_amdgcn_wmma_f32_16x16x32_f16(false, a1.v, false, bf.v, (short)0, ad[t],  false, false);
                    a2v[t] = __builtin_amdgcn_wmma_f32_16x16x32_f16(false, a2.v, false, bf.v, (short)0, a2v[t], false, false);
                }
            }

            #pragma unroll
            for (int t = 0; t < NBT; ++t) {
                const int col = (ng * NBT + t) * 16 + idx16;
                const float bv = bl[col];
                #pragma unroll
                for (int i = 0; i < 8; ++i) {
                    const int prow = p0 + hi * 8 + i;    // C/D layout: VGPR i -> row hi*8+i
                    float z   = az[t][i] + bv;
                    float dz  = ad[t][i];
                    float ddz = a2v[t][i];
                    float tt = fast_tanh(z);
                    float s  = 1.0f - tt * tt;
                    wbuf[(0 * PTS + prow) * ROWP + col] = (half_t)tt;
                    wbuf[(1 * PTS + prow) * ROWP + col] = (half_t)(s * dz);
                    wbuf[(2 * PTS + prow) * ROWP + col] = (half_t)(s * ddz - 2.0f * tt * s * dz * dz);
                }
            }
        }
    }
    __syncthreads();

    // ---- output layer: u'' = h3'' . W4 (bias has zero 2nd derivative) ----
    // final activations live in buffer 1 (after l = 2)
    if (tid < PTS) {
        const half_t* r = act + 1 * MATS + (2 * PTS + tid) * ROWP;
        float s = 0.0f;
        for (int c = 0; c < HID; c += 8) {
            v8h hv = *(const v8h*)(r + c);
            #pragma unroll
            for (int j = 0; j < 8; ++j) s += (float)hv[j] * W4[c + j];
        }
        out[pbase + tid] = s;
    }
}

// ---------------------------------------------------------------------------
// Boundary kernel: plain forward pass u(x), same tiling with a single matrix.
// ---------------------------------------------------------------------------
__global__ __launch_bounds__(THREADS) void pinn_bd_kernel(
    const float* __restrict__ x,
    const half_t* __restrict__ WT,
    const float* __restrict__ W0,
    const float* __restrict__ b0,
    const float* __restrict__ b1,
    const float* __restrict__ b2,
    const float* __restrict__ b3,
    const float* __restrict__ W4,
    const float* __restrict__ b4,
    float* __restrict__ out) {
    extern __shared__ char smem_raw[];
    half_t* act = (half_t*)smem_raw;           // [2 buf][PTS][ROWP]
    const int BUF = PTS * ROWP;

    const int tid   = threadIdx.x;
    const int pbase = blockIdx.x * PTS;

    for (int e = tid; e < PTS * HID; e += THREADS) {
        int p = e >> 8, c = e & 255;
        float z = x[pbase + p] * W0[c] + b0[c];
        act[p * ROWP + c] = (half_t)fast_tanh(z);
    }
    __syncthreads();

    const int lane  = tid & 31;
    const int idx16 = lane & 15;
    const int hi    = lane >> 4;
    const int p0    = (tid >> 5) * 16;

    #pragma unroll
    for (int l = 0; l < 3; ++l) {
        half_t* rbuf = act + (l & 1) * BUF;
        half_t* wbuf = act + ((l & 1) ^ 1) * BUF;
        const half_t* wt = WT + l * HID * HID;
        const float*  bl = (l == 0) ? b1 : ((l == 1) ? b2 : b3);

        for (int ng = 0; ng < 16 / NBT; ++ng) {
            v8f az[NBT];
            #pragma unroll
            for (int t = 0; t < NBT; ++t) az[t] = (v8f){0,0,0,0,0,0,0,0};

            for (int kb = 0; kb < 8; ++kb) {
                const int ko = kb * 32 + hi * 8;
                AFrag a0;
                const half_t* ar = rbuf + (p0 + idx16) * ROWP + ko;
                a0.h[0] = *(const v8h*)(ar);
                a0.h[1] = *(const v8h*)(ar + 16);
                #pragma unroll
                for (int t = 0; t < NBT; ++t) {
                    const int col = (ng * NBT + t) * 16 + idx16;
                    AFrag bf;
                    const half_t* br = wt + col * HID + ko;
                    bf.h[0] = *(const v8h*)(br);
                    bf.h[1] = *(const v8h*)(br + 16);
                    az[t] = __builtin_amdgcn_wmma_f32_16x16x32_f16(false, a0.v, false, bf.v, (short)0, az[t], false, false);
                }
            }

            #pragma unroll
            for (int t = 0; t < NBT; ++t) {
                const int col = (ng * NBT + t) * 16 + idx16;
                const float bv = bl[col];
                #pragma unroll
                for (int i = 0; i < 8; ++i) {
                    const int prow = p0 + hi * 8 + i;
                    wbuf[prow * ROWP + col] = (half_t)fast_tanh(az[t][i] + bv);
                }
            }
        }
    }
    __syncthreads();

    if (tid < PTS) {
        const half_t* r = act + 1 * BUF + tid * ROWP;   // final buffer = 1
        float s = 0.0f;
        for (int c = 0; c < HID; c += 8) {
            v8h hv = *(const v8h*)(r + c);
            #pragma unroll
            for (int j = 0; j < 8; ++j) s += (float)hv[j] * W4[c + j];
        }
        out[pbase + tid] = s + b4[0];
    }
}

// ---------------------------------------------------------------------------
extern "C" void kernel_launch(void* const* d_in, const int* in_sizes, int n_in,
                              void* d_out, int out_size, void* d_ws, size_t ws_size,
                              hipStream_t stream) {
    (void)in_sizes; (void)n_in; (void)out_size; (void)ws_size;
    const float* x_eq = (const float*)d_in[0];
    const float* x_bd = (const float*)d_in[1];
    const float* W0   = (const float*)d_in[2];
    const float* b0   = (const float*)d_in[3];
    const float* W1   = (const float*)d_in[4];
    const float* b1   = (const float*)d_in[5];
    const float* W2   = (const float*)d_in[6];
    const float* b2   = (const float*)d_in[7];
    const float* W3   = (const float*)d_in[8];
    const float* b3   = (const float*)d_in[9];
    const float* W4   = (const float*)d_in[10];
    const float* b4   = (const float*)d_in[11];

    half_t* WT = (half_t*)d_ws;                 // 3 * 256*256 f16 = 384 KB
    float* out_eq = (float*)d_out;              // [262144]
    float* out_bd = out_eq + 262144;            // [8192]

    wt_transpose_kernel<<<dim3(256, 3), 256, 0, stream>>>(W1, W2, W3, WT);

    const size_t lds_eq = (size_t)2 * 3 * PTS * ROWP * sizeof(half_t);  // ~198 KB
    const size_t lds_bd = (size_t)2 * PTS * ROWP * sizeof(half_t);      // ~66 KB
    (void)hipFuncSetAttribute((const void*)pinn_eq_kernel,
                              hipFuncAttributeMaxDynamicSharedMemorySize, (int)lds_eq);
    (void)hipFuncSetAttribute((const void*)pinn_bd_kernel,
                              hipFuncAttributeMaxDynamicSharedMemorySize, (int)lds_bd);

    pinn_eq_kernel<<<262144 / PTS, THREADS, lds_eq, stream>>>(
        x_eq, WT, W0, b0, b1, b2, b3, W4, out_eq);
    pinn_bd_kernel<<<8192 / PTS, THREADS, lds_bd, stream>>>(
        x_bd, WT, W0, b0, b1, b2, b3, W4, b4, out_bd);
}
